// SaMoEModel_Ab2_40278203302242
// MI455X (gfx1250) — matrix operand order repaired and checked by reading, not compiled, not run on hardware
//
#include <hip/hip_runtime.h>
#include <hip/hip_bf16.h>

// SaMoE fused kernel for gfx1250 (MI455X).
//   combined[n] = softmax_e( prior_logit[n,e] - err[n,e] ) . out[n,e]
// using the identity normalize(softmax(-err)*softmax(prior)) == softmax(prior - err).
//
// One wave32 processes a 16-row batch tile. The two H=32 reductions
// (expert einsum "enh,eh->en" and the gate GEMM hg @ gw2^T) run on
// v_wmma_f32_16x16x32_f16. Activations are computed in f32 VALU and packed
// to f16 per the CDNA5 A-matrix VGPR layout (CDNA5 ISA §7.12.2):
//   A (16x32 f16): lane L owns row M=L%16; halves [0..7]  = K = 8*hi + [0..7]
//                                          halves [8..15] = K = 16 + 8*hi + [0..7]
//   B (32x16 f16): lane L owns col N=L%16; halves [0..15] = same K order as A
//   C (16x16 f32): lane L owns col N=L%16; VGPR r = row M = r + 8*hi
// where hi = L>=16.

typedef __attribute__((ext_vector_type(16))) _Float16 v16h;
typedef __attribute__((ext_vector_type(8)))  float    v8f;

#define NE 8
#define NH 32

__global__ __launch_bounds__(256) void samoe_wmma_kernel(
    const float* __restrict__ ctx,  // (B, 8): even cols = cx, odd cols = cy
    const float* __restrict__ inp,  // (B, 1)
    const float* __restrict__ ew1, const float* __restrict__ eb1,  // (E,H)
    const float* __restrict__ ew2, const float* __restrict__ eb2,  // (E,H),(E)
    const float* __restrict__ gw1, const float* __restrict__ gb1,  // (H)
    const float* __restrict__ gw2, const float* __restrict__ gb2,  // (E,H),(E)
    float* __restrict__ out, int n)
{
    __shared__ float s_w1[NE * NH], s_b1[NE * NH], s_w2[NE * NH], s_gw2[NE * NH];
    __shared__ float s_gw1[NH], s_gb1[NH], s_eb2[NE], s_gb2[NE];

    const int tid = threadIdx.x;
    // Stage all weights into LDS once per block (1104 floats).
    s_w1[tid]  = ew1[tid];
    s_b1[tid]  = eb1[tid];
    s_w2[tid]  = ew2[tid];
    s_gw2[tid] = gw2[tid];
    if (tid < NH) { s_gw1[tid] = gw1[tid]; s_gb1[tid] = gb1[tid]; }
    if (tid < NE) { s_eb2[tid] = eb2[tid]; s_gb2[tid] = gb2[tid]; }
    __syncthreads();

    const int wave = tid >> 5;
    const int lane = tid & 31;
    const int tileBase = (blockIdx.x * 8 + wave) * 16;
    if (tileBase >= n) return;                 // wave-uniform: EXEC stays all-1s

    const int hi   = lane >> 4;                // which half of the wave
    const int nN   = lane & 15;                // B/C column (expert id when < 8)
    const int mrow = lane & 15;                // A-matrix row owned by this lane
    const int kbA  = hi * 8;                   // A k-base (plus +16 for upper run)
    const int kbB  = hi * 16;                  // unused (B is packed with kbA, like A)

    // The 5 scalar "x" streams for this lane's A-row: 4 context x's + input.
    const float4* crow = (const float4*)(ctx + (size_t)(tileBase + mrow) * 8);
    const float4 cA = crow[0], cB = crow[1];
    const float xs[5] = { cA.x, cA.z, cB.x, cB.z, inp[tileBase + mrow] };

    v8f acc[5] = {};   // pred accumulators per stream (C tiles, experts in cols 0..7)

    #pragma unroll
    for (int e = 0; e < NE; ++e) {
        // Per-lane slice of expert-e first-layer weights (A k-pattern).
        float w1v[16], b1v[16];
        #pragma unroll
        for (int i = 0; i < 8; ++i) {
            w1v[i]     = s_w1[e * NH + kbA + i];
            w1v[8 + i] = s_w1[e * NH + 16 + kbA + i];
            b1v[i]     = s_b1[e * NH + kbA + i];
            b1v[8 + i] = s_b1[e * NH + 16 + kbA + i];
        }
        // B matrix: only column e carries ew2[e,:]; other columns zero so the
        // 8 accumulating WMMAs fill disjoint expert columns of C.
        v16h bm;
        const bool sel = (nN == e);
        #pragma unroll
        for (int i = 0; i < 16; ++i)
            bm[i] = sel ? (_Float16)s_w2[e * NH + ((i < 8) ? (kbA + i) : (16 + kbA + i - 8))] : (_Float16)0.f;

        #pragma unroll
        for (int j = 0; j < 5; ++j) {
            v16h a;
            #pragma unroll
            for (int i = 0; i < 16; ++i)
                a[i] = (_Float16)fmaxf(fmaf(xs[j], w1v[i], b1v[i]), 0.f);
            acc[j] = __builtin_amdgcn_wmma_f32_16x16x32_f16(
                false, a, false, bm, (short)0, acc[j], false, false);
        }
    }

    // Prior gate: relu(input*gw1+gb1) @ gw2^T via one WMMA (experts in cols 0..7).
    v16h ga, gbm;
    #pragma unroll
    for (int i = 0; i < 8; ++i) {
        ga[i]     = (_Float16)fmaxf(fmaf(xs[4], s_gw1[kbA + i],      s_gb1[kbA + i]), 0.f);
        ga[8 + i] = (_Float16)fmaxf(fmaf(xs[4], s_gw1[16 + kbA + i], s_gb1[16 + kbA + i]), 0.f);
    }
    #pragma unroll
    for (int i = 0; i < 16; ++i)
        gbm[i] = (nN < NE) ? (_Float16)s_gw2[nN * NH + ((i < 8) ? (kbA + i) : (16 + kbA + i - 8))] : (_Float16)0.f;
    v8f gl = {};
    gl = __builtin_amdgcn_wmma_f32_16x16x32_f16(
        false, ga, false, gbm, (short)0, gl, false, false);

    const float e2 = (nN < NE) ? s_eb2[nN] : 0.f;
    const float g2 = (nN < NE) ? s_gb2[nN] : 0.f;

    // Epilogue: per C-row r, form s = prior_logit - err, fused softmax-combine
    // across the 16 lanes of each half (experts >= 8 masked to -inf / out=0).
    #pragma unroll
    for (int r = 0; r < 8; ++r) {
        const int row = tileBase + hi * 8 + r;
        const float4* rr = (const float4*)(ctx + (size_t)row * 8);
        const float4 r0 = rr[0], r1 = rr[1];
        const float d0 = acc[0][r] + e2 - r0.y;
        const float d1 = acc[1][r] + e2 - r0.w;
        const float d2 = acc[2][r] + e2 - r1.y;
        const float d3 = acc[3][r] + e2 - r1.w;
        const float err  = d0 * d0 + d1 * d1 + d2 * d2 + d3 * d3;
        const float outv = acc[4][r] + e2;
        const float s = (nN < NE) ? (gl[r] + g2 - err) : -1e30f;

        float mx = s;
        #pragma unroll
        for (int msk = 1; msk <= 8; msk <<= 1)
            mx = fmaxf(mx, __shfl_xor(mx, msk, 32));
        const float ex = __expf(s - mx);
        float num = ex * outv;
        float den = ex;
        #pragma unroll
        for (int msk = 1; msk <= 8; msk <<= 1) {
            num += __shfl_xor(num, msk, 32);
            den += __shfl_xor(den, msk, 32);
        }
        if ((lane & 15) == 0)
            out[row] = num / den;
    }
}

extern "C" void kernel_launch(void* const* d_in, const int* in_sizes, int n_in,
                              void* d_out, int out_size, void* d_ws, size_t ws_size,
                              hipStream_t stream) {
    const float* ctx = (const float*)d_in[0];
    const float* inp = (const float*)d_in[1];
    const float* ew1 = (const float*)d_in[2];
    const float* eb1 = (const float*)d_in[3];
    const float* ew2 = (const float*)d_in[4];
    const float* eb2 = (const float*)d_in[5];
    const float* gw1 = (const float*)d_in[6];
    const float* gb1 = (const float*)d_in[7];
    const float* gw2 = (const float*)d_in[8];
    const float* gb2 = (const float*)d_in[9];
    float* outp = (float*)d_out;

    const int n = in_sizes[1];                 // batch size B (input is (B,1))
    const int rowsPerBlock = 16 * 8;           // 8 waves/block, 16-row tile each
    const int blocks = (n + rowsPerBlock - 1) / rowsPerBlock;
    samoe_wmma_kernel<<<blocks, 256, 0, stream>>>(
        ctx, inp, ew1, eb1, ew2, eb2, gw1, gb1, gw2, gb2, outp, n);
}
